// STCAOutputLayer_46316927320613
// MI455X (gfx1250) — matrix operand order repaired and checked by reading, not compiled, not run on hardware
//
#include <hip/hip_runtime.h>
#include <hip/hip_bf16.h>
#include <stdint.h>

typedef __attribute__((ext_vector_type(16))) __bf16 v16bf;
typedef __attribute__((ext_vector_type(8)))  float  v8f;

#define B_DIM 32
#define T_DIM 500
#define K_DIM 2048
#define N_DIM 1024
#define M_DIM (B_DIM * T_DIM)   // 16000

#define BM 128
#define BN 64
#define BK 32
#define LDA 36                  // LDS row stride (ushorts) for A tiles (pad vs bank conflicts)
#define LDB 40                  // LDS row stride (ushorts) for B tiles: 80B -> every 16B chunk aligned
#define KSTEPS (K_DIM / BK)     // 64
#define EPS_F 1e-8f

union FragBF { unsigned short u[16]; v16bf v; };

__device__ __forceinline__ void bf16split(float a, unsigned short& hi, unsigned short& lo) {
  unsigned int ua = __builtin_bit_cast(unsigned int, a);
  hi = (unsigned short)(ua >> 16);
  float fhi = __builtin_bit_cast(float, ua & 0xFFFF0000u);
  float r = a - fhi;                         // exact
  lo = (unsigned short)(__builtin_bit_cast(unsigned int, r) >> 16);
}

// ---------------------------------------------------------------------------
// Kernel 1: split w into bf16 hi/lo planes stored TRANSPOSED [N][K] (so GEMM
// B-tiles are contiguous K-runs -> async-copyable), plus per-column norms.
// ---------------------------------------------------------------------------
__global__ void __launch_bounds__(256) prep_w(const float* __restrict__ w,
                                              unsigned short* __restrict__ whiT,
                                              unsigned short* __restrict__ wloT,
                                              float* __restrict__ nrm) {
  __shared__ float sred[8][32];
  int tid = threadIdx.x;
  int ol = tid & 31;
  int sl = tid >> 5;                         // K-slice 0..7
  int o  = blockIdx.x * 32 + ol;
  float acc = 0.f;
  for (int j = 0; j < K_DIM / 8; ++j) {
    int c = sl * (K_DIM / 8) + j;
    float v = w[(size_t)c * N_DIM + o];      // coalesced read
    unsigned short h, l;
    bf16split(v, h, l);
    size_t tidx = (size_t)o * K_DIM + c;     // transposed store
    whiT[tidx] = h;
    wloT[tidx] = l;
    acc += v * v;
  }
  sred[sl][ol] = acc;
  __syncthreads();
  if (sl == 0) {
    float s = 0.f;
#pragma unroll
    for (int i = 0; i < 8; ++i) s += sred[i][ol];
    nrm[o] = s;
  }
}

// ---------------------------------------------------------------------------
// GEMM helpers
// ---------------------------------------------------------------------------
__device__ __forceinline__ void stage_B_async(const unsigned short* __restrict__ whiT,
                                              const unsigned short* __restrict__ wloT,
                                              unsigned short (*bsh)[LDB],
                                              unsigned short (*bsl)[LDB],
                                              int n0, int k0, int tid) {
  int nn = tid >> 2;                 // 0..63
  int ch = (tid & 3) * 8;            // ushort offset: 0,8,16,24  (16B chunks)
  size_t gi = (size_t)(n0 + nn) * K_DIM + k0 + ch;
  unsigned ldsH = (unsigned)(uintptr_t)&bsh[nn][ch];
  unsigned ldsL = (unsigned)(uintptr_t)&bsl[nn][ch];
  unsigned long long gaH = (unsigned long long)(uintptr_t)(whiT + gi);
  unsigned long long gaL = (unsigned long long)(uintptr_t)(wloT + gi);
  asm volatile("global_load_async_to_lds_b128 %0, %1, off" :: "v"(ldsH), "v"(gaH) : "memory");
  asm volatile("global_load_async_to_lds_b128 %0, %1, off" :: "v"(ldsL), "v"(gaL) : "memory");
}

__device__ __forceinline__ void load_x_regs(const float* __restrict__ x,
                                            int m0, int k0, int tid, float4 xv[4]) {
#pragma unroll
  for (int i = 0; i < 4; ++i) {
    int e  = tid + i * 256;          // 0..1023 float4s of the 128x32 tile
    int r  = e >> 3;
    int c4 = (e & 7) * 4;
    xv[i] = *reinterpret_cast<const float4*>(&x[(size_t)(m0 + r) * K_DIM + k0 + c4]);
  }
}

__device__ __forceinline__ void store_x_lds(const float4 xv[4],
                                            unsigned short (*ash)[LDA],
                                            unsigned short (*asl)[LDA], int tid) {
#pragma unroll
  for (int i = 0; i < 4; ++i) {
    int e  = tid + i * 256;
    int r  = e >> 3;
    int c4 = (e & 7) * 4;
    unsigned short h, l;
    bf16split(xv[i].x, h, l); ash[r][c4 + 0] = h; asl[r][c4 + 0] = l;
    bf16split(xv[i].y, h, l); ash[r][c4 + 1] = h; asl[r][c4 + 1] = l;
    bf16split(xv[i].z, h, l); ash[r][c4 + 2] = h; asl[r][c4 + 2] = l;
    bf16split(xv[i].w, h, l); ash[r][c4 + 3] = h; asl[r][c4 + 3] = l;
  }
}

// ---------------------------------------------------------------------------
// Kernel 2: h = x @ w via bf16x3 WMMA, double-buffered LDS, async B staging.
// Block tile 128x64, 8 waves, each wave owns a 32x32 region (2x2 16x16 tiles).
// ---------------------------------------------------------------------------
__global__ void __launch_bounds__(256) gemm_bf16x3(const float* __restrict__ x,
                                                   const unsigned short* __restrict__ whiT,
                                                   const unsigned short* __restrict__ wloT,
                                                   float* __restrict__ hout) {
  __shared__ __attribute__((aligned(16))) unsigned short AsH[2][BM][LDA];
  __shared__ __attribute__((aligned(16))) unsigned short AsL[2][BM][LDA];
  __shared__ __attribute__((aligned(16))) unsigned short BsH[2][BN][LDB];
  __shared__ __attribute__((aligned(16))) unsigned short BsL[2][BN][LDB];

  int tid   = threadIdx.x;
  int lane  = tid & 31;
  int wv    = tid >> 5;
  int waveM = wv & 3;            // rows waveM*32
  int waveN = wv >> 2;           // cols waveN*32
  int m0 = blockIdx.y * BM;
  int n0 = blockIdx.x * BN;

  int halfsel = lane >> 4;
  int l15     = lane & 15;

  v8f acc[2][2] = {};

  // ---- prologue: stage tile 0 into buffer 0 ----
  {
    stage_B_async(whiT, wloT, BsH[0], BsL[0], n0, /*k0=*/0, tid);
    float4 xv[4];
    load_x_regs(x, m0, 0, tid, xv);
    store_x_lds(xv, AsH[0], AsL[0], tid);
  }
  asm volatile("s_wait_asynccnt 0x0" ::: "memory");
  __syncthreads();

  for (int kt = 0; kt < KSTEPS; ++kt) {
    int p  = kt & 1;
    int np = p ^ 1;
    bool more = (kt + 1) < KSTEPS;

    // ---- issue next tile's async B copy + pull next x tile into registers ----
    float4 xv[4];
    if (more) {
      stage_B_async(whiT, wloT, BsH[np], BsL[np], n0, (kt + 1) * BK, tid);
      load_x_regs(x, m0, (kt + 1) * BK, tid, xv);
    }

    // ---- fragments from buffer p (ISA 16-bit A/B layouts) ----
    FragBF ah[2], al[2], bh[2], bl[2];
#pragma unroll
    for (int mt = 0; mt < 2; ++mt) {
      const unsigned short* pah = &AsH[p][waveM * 32 + mt * 16 + l15][halfsel * 8];
      const unsigned short* pal = &AsL[p][waveM * 32 + mt * 16 + l15][halfsel * 8];
#pragma unroll
      for (int j = 0; j < 8; ++j) {
        ah[mt].u[j]     = pah[j];
        ah[mt].u[8 + j] = pah[16 + j];
        al[mt].u[j]     = pal[j];
        al[mt].u[8 + j] = pal[16 + j];
      }
    }
#pragma unroll
    for (int nt = 0; nt < 2; ++nt) {
      const unsigned short* pbh = &BsH[p][waveN * 32 + nt * 16 + l15][halfsel * 8];
      const unsigned short* pbl = &BsL[p][waveN * 32 + nt * 16 + l15][halfsel * 8];
#pragma unroll
      for (int j = 0; j < 8; ++j) {
        bh[nt].u[j]     = pbh[j];
        bh[nt].u[8 + j] = pbh[16 + j];
        bl[nt].u[j]     = pbl[j];
        bl[nt].u[8 + j] = pbl[16 + j];
      }
    }

    // ---- 12 WMMAs: hi*hi + hi*lo + lo*hi per 16x16 subtile ----
#pragma unroll
    for (int mt = 0; mt < 2; ++mt)
#pragma unroll
      for (int nt = 0; nt < 2; ++nt) {
        acc[mt][nt] = __builtin_amdgcn_wmma_f32_16x16x32_bf16(
            false, ah[mt].v, false, bh[nt].v, (short)0, acc[mt][nt], false, false);
        acc[mt][nt] = __builtin_amdgcn_wmma_f32_16x16x32_bf16(
            false, ah[mt].v, false, bl[nt].v, (short)0, acc[mt][nt], false, false);
        acc[mt][nt] = __builtin_amdgcn_wmma_f32_16x16x32_bf16(
            false, al[mt].v, false, bh[nt].v, (short)0, acc[mt][nt], false, false);
      }

    // ---- convert+park next x tile into buffer np (was idle since kt-1 barrier) ----
    if (more) {
      store_x_lds(xv, AsH[np], AsL[np], tid);
      asm volatile("s_wait_asynccnt 0x0" ::: "memory");
    }
    __syncthreads();
  }

  // ---- epilogue: C layout -> global (lanes 0-15: M=v, lanes 16-31: M=v+8) ----
#pragma unroll
  for (int mt = 0; mt < 2; ++mt)
#pragma unroll
    for (int nt = 0; nt < 2; ++nt) {
      int row0 = m0 + waveM * 32 + mt * 16 + halfsel * 8;
      int col  = n0 + waveN * 32 + nt * 16 + l15;
#pragma unroll
      for (int v = 0; v < 8; ++v)
        hout[(size_t)(row0 + v) * N_DIM + col] = acc[mt][nt][v];
    }
}

// ---------------------------------------------------------------------------
// Kernel 3: per-(b,o) temporal scan in place over h (= d_out).
// E stays identically zero in the reference, so mem = M - S.
// ---------------------------------------------------------------------------
__global__ void __launch_bounds__(256) scan_kernel(float* __restrict__ h,
                                                   const float* __restrict__ nrm,
                                                   const float* __restrict__ bias,
                                                   const float* __restrict__ dmp,
                                                   const float* __restrict__ dsp,
                                                   float* __restrict__ partials) {
  __shared__ float red[256];
  int tid = threadIdx.x;
  int idx = blockIdx.x * 256 + tid;
  int o   = idx & (N_DIM - 1);
  int bb  = idx >> 10;
  float invn = 1.0f / (nrm[o] + EPS_F);
  float bo   = bias[o];
  float dm   = dmp[0];
  float dsv  = dsp[0];
  float Mv = 0.f, Sv = 0.f, ss = 0.f;
  float* hp = h + (size_t)bb * T_DIM * N_DIM + o;
  for (int t = 0; t < T_DIM; ++t) {
    float hv = hp[(size_t)t * N_DIM];
    Mv = dm * (Mv + hv);
    Sv = dsv * (Sv + hv);
    float out = (Mv - Sv) * invn - bo;
    hp[(size_t)t * N_DIM] = out;
    ss += out * out;
  }
  red[tid] = ss;
  __syncthreads();
  for (int s = 128; s > 0; s >>= 1) {
    if (tid < s) red[tid] += red[tid + s];
    __syncthreads();
  }
  if (tid == 0) partials[blockIdx.x] = red[0];
}

// ---------------------------------------------------------------------------
// Kernel 4: deterministic final reduction -> loss = 0.5 * mean(out^2)
// ---------------------------------------------------------------------------
__global__ void __launch_bounds__(128) loss_kernel(const float* __restrict__ partials,
                                                   float* __restrict__ out_loss) {
  __shared__ float red[128];
  int tid = threadIdx.x;
  red[tid] = partials[tid];
  __syncthreads();
  for (int s = 64; s > 0; s >>= 1) {
    if (tid < s) red[tid] += red[tid + s];
    __syncthreads();
  }
  if (tid == 0) out_loss[0] = 0.5f * red[0] / (float)((size_t)M_DIM * (size_t)N_DIM);
}

// ---------------------------------------------------------------------------
extern "C" void kernel_launch(void* const* d_in, const int* in_sizes, int n_in,
                              void* d_out, int out_size, void* d_ws, size_t ws_size,
                              hipStream_t stream) {
  const float* x   = (const float*)d_in[0];   // [32,500,2048]
  const float* w   = (const float*)d_in[1];   // [2048,1024]
  const float* b   = (const float*)d_in[2];   // [1024]
  const float* dm  = (const float*)d_in[3];   // [1]
  const float* dsv = (const float*)d_in[4];   // [1]
  float* out = (float*)d_out;                 // [16000*1024] spk_rec, then 1 loss

  // workspace: w_hiT (4MB) | w_loT (4MB) | norms (4KB) | partial sums (512B)
  unsigned short* whiT = (unsigned short*)d_ws;
  unsigned short* wloT = whiT + (size_t)K_DIM * N_DIM;
  float* nrm      = (float*)(wloT + (size_t)K_DIM * N_DIM);
  float* partials = nrm + N_DIM;

  prep_w<<<N_DIM / 32, 256, 0, stream>>>(w, whiT, wloT, nrm);

  dim3 grid(N_DIM / BN, M_DIM / BM);          // 16 x 125
  gemm_bf16x3<<<grid, 256, 0, stream>>>(x, whiT, wloT, out);

  scan_kernel<<<(B_DIM * N_DIM) / 256, 256, 0, stream>>>(out, nrm, b, dm, dsv, partials);

  loss_kernel<<<1, 128, 0, stream>>>(partials, out + (size_t)M_DIM * N_DIM);
}
